// SpatialAttention_7146825580586
// MI455X (gfx1250) — matrix operand order repaired and checked by reading, not compiled, not run on hardware
//
#include <hip/hip_runtime.h>
#include <hip/hip_bf16.h>

typedef __attribute__((ext_vector_type(16))) __bf16 v16bf;
typedef __attribute__((ext_vector_type(8)))  __bf16 v8bf;
typedef __attribute__((ext_vector_type(8)))  float  v8f;

#define B_ 4
#define C_ 64
#define N_ 4096

__device__ __forceinline__ int a_kidx(int e, int hf) {
    // 16-bit A-matrix 16x32 layout: lanes 0-15 hold K={0..7,16..23}, lanes 16-31 K={8..15,24..31}
    return ((e < 8) ? 0 : 16) + hf * 8 + (e & 7);
}

__device__ __forceinline__ v8f bf16_wmma(v16bf a, v16bf b, v8f c) {
    return __builtin_amdgcn_wmma_f32_16x16x32_bf16(false, a, false, b, (short)0, c, false, false);
}

// ---------------------------------------------------------------------------
// Stage 1: q/k/v = W*x + b (bf16).
//   q,k written TRANSPOSED  [B][N][C]  (stage-2 A-/K-fragment loads contiguous;
//                                       also makes the store here one b128!)
//   v   written row-major   [B][C][N]  (stage-2 V-fragment loads contiguous)
// Grid: (N/64, B), 256 threads (8 waves); 48 16x16 tiles, K=64 -> 2 WMMA each.
// ---------------------------------------------------------------------------
__global__ void __launch_bounds__(256)
qkv_project_kernel(const float* __restrict__ x,
                   const float* __restrict__ Wq, const float* __restrict__ bq,
                   const float* __restrict__ Wk, const float* __restrict__ bk,
                   const float* __restrict__ Wv, const float* __restrict__ bv,
                   __bf16* __restrict__ qT, __bf16* __restrict__ kT,
                   __bf16* __restrict__ vN)
{
    __shared__ __bf16 wlds[3 * 64 * 64];
    __shared__ __bf16 xlds[64 * 64];
    __shared__ float  blds[3 * 64];

    const int b  = blockIdx.y;
    const int n0 = blockIdx.x * 64;
    const int tid = threadIdx.x;

    for (int idx = tid; idx < 3 * 64 * 64; idx += 256) {
        const float* Wsrc = (idx < 4096) ? Wq : (idx < 8192) ? Wk : Wv;
        wlds[idx] = (__bf16)Wsrc[idx & 4095];
    }
    for (int idx = tid; idx < 192; idx += 256) {
        const float* bsrc = (idx < 64) ? bq : (idx < 128) ? bk : bv;
        blds[idx] = bsrc[idx & 63];
    }
    for (int idx = tid; idx < 64 * 64; idx += 256) {
        const int c = idx >> 6, n = idx & 63;
        xlds[idx] = (__bf16)x[((size_t)b * C_ + c) * N_ + n0 + n];
    }
    __syncthreads();

    const int wave = tid >> 5;
    const int lane = tid & 31;
    const int li = lane & 15;
    const int hf = lane >> 4;

    for (int t = wave; t < 48; t += 8) {
        const int mat = t >> 4;      // 0=q 1=k 2=v
        const int rem = t & 15;
        const int ot  = rem >> 2;    // out-channel tile
        const int nt  = rem & 3;     // n tile
        const __bf16* W = wlds + mat * 4096;

        v16bf a0, a1, m0, m1;
        #pragma unroll
        for (int e = 0; e < 16; ++e) {
            const int kk = hf * 16 + e;
            a0[e] = W[(ot * 16 + li) * 64 + kk];        // A: W rows o, K=c 0..31
            a1[e] = W[(ot * 16 + li) * 64 + 32 + kk];   // K=c 32..63
            m0[e] = xlds[(hf * 16 + e) * 64 + nt * 16 + li];        // B: K=c row, col n
            m1[e] = xlds[(32 + hf * 16 + e) * 64 + nt * 16 + li];
        }
        v8f acc = {};
        acc = bf16_wmma(a0, m0, acc);
        acc = bf16_wmma(a1, m1, acc);

        const int n = n0 + nt * 16 + li;            // D col = spatial index
        if (mat < 2) {
            // Lane owns 8 CONSECUTIVE channels of one n-row -> single 16B store.
            __bf16* dstp = ((mat == 0) ? qT : kT) + (size_t)b * N_ * C_;
            v8bf pk;
            #pragma unroll
            for (int r = 0; r < 8; ++r)
                pk[r] = (__bf16)(acc[r] + blds[mat * 64 + ot * 16 + 8 * hf + r]);
            *(v8bf*)&dstp[(size_t)n * C_ + ot * 16 + 8 * hf] = pk;
        } else {
            // v row-major [c][n]: channel-strided scatter (8 x b16).
            __bf16* dstp = vN + (size_t)b * C_ * N_;
            #pragma unroll
            for (int r = 0; r < 8; ++r) {
                const int o = ot * 16 + 8 * hf + r;
                dstp[(size_t)o * N_ + n] = (__bf16)(acc[r] + blds[2 * 64 + o]);
            }
        }
    }
}

// ---------------------------------------------------------------------------
// Stage 2: fused flash attention + residual, double-buffered async-to-LDS K/V.
//   energy[i,j] = sum_c qT[i,c]*kT[j,c]   (online softmax over j)
//   outT[i,c]   = sum_j P[i,j]*vN[c,j]
//   out[b,c,i]  = gamma*outT[i,c]/l_i + x[b,c,i]
// Grid: (N/128, B), 256 threads (8 waves x 16 query rows).
// ---------------------------------------------------------------------------
__global__ void __launch_bounds__(256)
flash_attn_kernel(const __bf16* __restrict__ qT, const __bf16* __restrict__ kT,
                  const __bf16* __restrict__ vN, const float* __restrict__ x,
                  const float* __restrict__ gamma, float* __restrict__ out)
{
    __shared__ __bf16 kbuf[2][32 * 64];   // [j 32][c 64]
    __shared__ __bf16 vbuf[2][64 * 32];   // [c 64][j 32]
    __shared__ __bf16 pbuf[8][16 * 32];

    const int b    = blockIdx.y;
    const int i0   = blockIdx.x * 128;
    const int tid  = threadIdx.x;
    const int wave = tid >> 5;
    const int lane = tid & 31;
    const int li   = lane & 15;
    const int hf   = lane >> 4;
    const int iw   = i0 + wave * 16;     // this wave's 16 query rows

    const __bf16* kbase = kT + (size_t)b * N_ * C_;
    const __bf16* vbase = vN + (size_t)b * C_ * N_;

    // Issue async global->LDS staging for j block (32 rows of K, 32 cols of V).
    // One b128 (16B) per thread per matrix; tracked on ASYNCcnt.
    auto issue_stage = [&](int j) {
        const int buf = (j >> 5) & 1;
        // K block: kT rows j..j+31 are 2048 contiguous halves.
        const unsigned kls = (unsigned)(size_t)&kbuf[buf][tid * 8];
        const unsigned kgo = (unsigned)(j * C_ + tid * 8) * 2u;
        asm volatile("global_load_async_to_lds_b128 %0, %1, %2"
                     :: "v"(kls), "v"(kgo), "s"(kbase) : "memory");
        // V block: 64 rows (c) x 32 halves (j), row stride N_.
        const int row = tid >> 2, ch = tid & 3;
        const unsigned vls = (unsigned)(size_t)&vbuf[buf][row * 32 + ch * 8];
        const unsigned vgo = (unsigned)(row * N_ + j + ch * 8) * 2u;
        asm volatile("global_load_async_to_lds_b128 %0, %1, %2"
                     :: "v"(vls), "v"(vgo), "s"(vbase) : "memory");
    };

    // Persistent A fragments for Q (rows i, K = channel 0..63)
    v16bf aq0, aq1;
    {
        const __bf16* qrow = qT + ((size_t)b * N_ + iw + li) * C_;
        #pragma unroll
        for (int e = 0; e < 16; ++e) {
            const int kk = hf * 16 + e;
            aq0[e] = qrow[kk];
            aq1[e] = qrow[32 + kk];
        }
    }

    v8f acc0 = {}, acc1 = {}, acc2 = {}, acc3 = {};   // outT tiles: cols c 0..63
    float mrow[8], lrow[8];
    #pragma unroll
    for (int r = 0; r < 8; ++r) { mrow[r] = -1.0e30f; lrow[r] = 0.0f; }

    issue_stage(0);   // prologue prefetch

    for (int j0 = 0; j0 < N_; j0 += 32) {
        const int cur = (j0 >> 5) & 1;
        // Drain my async loads; barrier publishes the tile block-wide and
        // proves everyone finished reading the other (prefetch-target) buffer.
        asm volatile("s_wait_asynccnt 0x0" ::: "memory");
        __syncthreads();
        if (j0 + 32 < N_) issue_stage(j0 + 32);   // overlaps with compute below

        const __bf16* kb = kbuf[cur];
        const __bf16* vb = vbuf[cur];

        // B fragments for K: col j = li (+16), K = channel (contiguous 16 halves)
        v16bf bk00 = *(const v16bf*)&kb[(li) * 64 + hf * 16];
        v16bf bk01 = *(const v16bf*)&kb[(li) * 64 + 32 + hf * 16];
        v16bf bk10 = *(const v16bf*)&kb[(16 + li) * 64 + hf * 16];
        v16bf bk11 = *(const v16bf*)&kb[(16 + li) * 64 + 32 + hf * 16];

        v8f s0 = {}, s1 = {};
        s0 = bf16_wmma(aq0, bk00, s0);
        s0 = bf16_wmma(aq1, bk01, s0);
        s1 = bf16_wmma(aq0, bk10, s1);
        s1 = bf16_wmma(aq1, bk11, s1);

        // Online softmax over this 32-wide j block. Row i = r + 8*hf is shared
        // by the 16 lanes of one wave-half -> xor-shuffle reduce with masks<16.
        #pragma unroll
        for (int r = 0; r < 8; ++r) {
            const float v0 = s0[r], v1 = s1[r];
            float mx = fmaxf(v0, v1);
            mx = fmaxf(mx, __shfl_xor(mx, 1, 32));
            mx = fmaxf(mx, __shfl_xor(mx, 2, 32));
            mx = fmaxf(mx, __shfl_xor(mx, 4, 32));
            mx = fmaxf(mx, __shfl_xor(mx, 8, 32));
            const float mnew  = fmaxf(mrow[r], mx);
            const float scale = __expf(mrow[r] - mnew);
            const float p0 = __expf(v0 - mnew);
            const float p1 = __expf(v1 - mnew);
            float rs = p0 + p1;
            rs += __shfl_xor(rs, 1, 32);
            rs += __shfl_xor(rs, 2, 32);
            rs += __shfl_xor(rs, 4, 32);
            rs += __shfl_xor(rs, 8, 32);
            lrow[r] = lrow[r] * scale + rs;
            mrow[r] = mnew;
            acc0[r] *= scale; acc1[r] *= scale; acc2[r] *= scale; acc3[r] *= scale;
            // Stash P (D-layout) into per-wave LDS for transpose to A-layout
            pbuf[wave][(r + 8 * hf) * 32 + li]      = (__bf16)p0;
            pbuf[wave][(r + 8 * hf) * 32 + 16 + li] = (__bf16)p1;
        }

        // Intra-wave LDS RAW: DS ops are in-order per wave; make the wait explicit.
        asm volatile("s_wait_dscnt 0" ::: "memory");

        v16bf pa;
        #pragma unroll
        for (int e = 0; e < 16; ++e)
            pa[e] = pbuf[wave][li * 32 + hf * 16 + e];   // A: row i = li, K = j

        // B fragments for V: col c = ct*16+li, K = j (contiguous 16 halves)
        v16bf bv0 = *(const v16bf*)&vb[(li) * 32 + hf * 16];
        v16bf bv1 = *(const v16bf*)&vb[(16 + li) * 32 + hf * 16];
        v16bf bv2 = *(const v16bf*)&vb[(32 + li) * 32 + hf * 16];
        v16bf bv3 = *(const v16bf*)&vb[(48 + li) * 32 + hf * 16];

        acc0 = bf16_wmma(pa, bv0, acc0);
        acc1 = bf16_wmma(pa, bv1, acc1);
        acc2 = bf16_wmma(pa, bv2, acc2);
        acc3 = bf16_wmma(pa, bv3, acc3);
    }

    const float g = gamma[0];
    #pragma unroll
    for (int r = 0; r < 8; ++r) {
        const float inv = 1.0f / lrow[r];
        const int i = iw + r + 8 * hf;          // D row = query index
        #pragma unroll
        for (int ct = 0; ct < 4; ++ct) {
            const int c = ct * 16 + li;         // D col = channel
            const float a = (ct == 0) ? acc0[r] : (ct == 1) ? acc1[r]
                          : (ct == 2) ? acc2[r] : acc3[r];
            const size_t idx = ((size_t)b * C_ + c) * N_ + i;
            out[idx] = g * (a * inv) + x[idx];
        }
    }
}

extern "C" void kernel_launch(void* const* d_in, const int* in_sizes, int n_in,
                              void* d_out, int out_size, void* d_ws, size_t ws_size,
                              hipStream_t stream) {
    const float* x     = (const float*)d_in[0];
    const float* Wq    = (const float*)d_in[1];
    const float* bq    = (const float*)d_in[2];
    const float* Wk    = (const float*)d_in[3];
    const float* bk    = (const float*)d_in[4];
    const float* Wv    = (const float*)d_in[5];
    const float* bv    = (const float*)d_in[6];
    const float* gamma = (const float*)d_in[7];
    float* out = (float*)d_out;

    // Workspace: qT/kT [B][N][C], vN [B][C][N], bf16; 2 MiB each (6 MiB total).
    __bf16* qT = (__bf16*)d_ws;
    __bf16* kT = qT + (size_t)B_ * N_ * C_;
    __bf16* vN = kT + (size_t)B_ * N_ * C_;

    qkv_project_kernel<<<dim3(N_ / 64, B_), 256, 0, stream>>>(
        x, Wq, bq, Wk, bk, Wv, bv, qT, kT, vN);
    flash_attn_kernel<<<dim3(N_ / 128, B_), 256, 0, stream>>>(
        qT, kT, vN, x, gamma, out);
}
